// EMPSNLayer_14903536517824
// MI455X (gfx1250) — compile-verified, matrix operand order checked by reading, not run on hardware
//
#include <hip/hip_runtime.h>

typedef __bf16 bf16_t;
typedef bf16_t v16bf __attribute__((ext_vector_type(16)));
typedef bf16_t v8bf  __attribute__((ext_vector_type(8)));
typedef float  v8f   __attribute__((ext_vector_type(8)));
typedef float  v4f   __attribute__((ext_vector_type(4)));

#define WAVES_PER_BLOCK 8
#define TPB (WAVES_PER_BLOCK * 32)
#define MAX_BLOCKS 1024

__device__ __forceinline__ float sigm(float x) { return 1.0f / (1.0f + __expf(-x)); }
__device__ __forceinline__ float act_apply(float x, int a) {
  if (a == 1) return sigm(x);          // sigmoid
  if (a == 2) return x * sigm(x);      // silu
  return x;                            // identity
}
__device__ __forceinline__ v8f vzero8() {
  v8f v;
#pragma unroll
  for (int i = 0; i < 8; ++i) v[i] = 0.0f;
  return v;
}

// K-index inside a 32-wide chunk for element e of a 16-bit A/B fragment.
__device__ __forceinline__ int klocal(int e, int half) {
  return ((e < 8) ? e : e + 8) + half * 8;
}

// 32-byte (16 x bf16) LDS fragment load as 2x ds_load_b128.
__device__ __forceinline__ v16bf ldsfrag16(const bf16_t* p) {
  const v8bf* q = (const v8bf*)p;
  v8bf lo = q[0];
  v8bf hi = q[1];
  return __builtin_shufflevector(lo, hi, 0, 1, 2, 3, 4, 5, 6, 7,
                                 8, 9, 10, 11, 12, 13, 14, 15);
}

// Pre-swizzle a [din x 64] f32 weight into fragment-major bf16 LDS layout:
// dstf[((kc*4+nt)*32 + lane)*16 + e] = W[(kc*32+klocal(e,half))*64 + nt*16+(lane&15)]
// Rows k >= din are zero-padded.
__device__ __forceinline__ void swizzle_weights(bf16_t* dstf, const float* __restrict__ W,
                                                int nchunks, int din, int tid) {
  for (int f = tid; f < nchunks * 4 * 32; f += TPB) {
    const int lane_f = f & 31;
    const int nt = (f >> 5) & 3;
    const int kc = f >> 7;
    const int half_f = lane_f >> 4;
    const int n = (lane_f & 15) + nt * 16;
    bf16_t* d = dstf + f * 16;
#pragma unroll
    for (int e = 0; e < 16; ++e) {
      int k = kc * 32 + klocal(e, half_f);
      d[e] = (bf16_t)((k < din) ? W[k * 64 + n] : 0.0f);
    }
  }
}

// B fragment from fragment-major LDS weights.
__device__ __forceinline__ v16bf bfrag(const bf16_t* wf, int kc, int nt, int lane) {
  return ldsfrag16(wf + (size_t)((kc * 4 + nt) * 32 + lane) * 16);
}

// A fragment from an aligned 64-float global row (chunks fully inside the row).
__device__ __forceinline__ v16bf afrag_row64(const float* __restrict__ p, int off, int half) {
  const v4f* q = (const v4f*)(p + off + half * 8);
  v4f f0 = q[0], f1 = q[1], f2 = q[4], f3 = q[5];
  v16bf a;
#pragma unroll
  for (int j = 0; j < 4; ++j) {
    a[j]      = (bf16_t)f0[j];
    a[4 + j]  = (bf16_t)f1[j];
    a[8 + j]  = (bf16_t)f2[j];
    a[12 + j] = (bf16_t)f3[j];
  }
  return a;
}

// A fragment for the (short, unaligned) invariant chunk; zero-padded past nvalid.
__device__ __forceinline__ v16bf afrag_inv(const float* __restrict__ p, int nvalid, int half) {
  v16bf a;
#pragma unroll
  for (int e = 0; e < 16; ++e) {
    int k = klocal(e, half);
    float v = (k < nvalid) ? p[k] : 0.0f;
    a[e] = (bf16_t)v;
  }
  return a;
}

// A fragment from the per-wave row-major [16 x 64] bf16 staging buffer.
__device__ __forceinline__ v16bf afrag_stage(const bf16_t* m, int kc, int lane) {
  const int row = lane & 15, half = lane >> 4;
  const bf16_t* base = m + row * 64 + kc * 32 + half * 8;
  v8bf lo = *(const v8bf*)base;
  v8bf hi = *(const v8bf*)(base + 16);
  return __builtin_shufflevector(lo, hi, 0, 1, 2, 3, 4, 5, 6, 7,
                                 8, 9, 10, 11, 12, 13, 14, 15);
}

#define WMMA_BF16(a, b, c) \
  __builtin_amdgcn_wmma_f32_16x16x32_bf16(false, (a), false, (b), (short)0, (c), false, false)

// ---------------------------------------------------------------------------
// Edge-message kernel: per edge e
//   h = [x_send[sidx[e]], x_recv[ridx[e]], inv[e]]          (K = 128+ninv -> 160)
//   m = act2(act1(h@W1+b1)@W2+b2);  gate = sigmoid(m@W3+b3)
//   atomicAdd(agg[ridx[e]], m*gate)
// One wave handles a 16-edge tile (M=16), N=64 via 4 tiles of 16.
// ---------------------------------------------------------------------------
__global__ __launch_bounds__(TPB) void edge_msg_kernel(
    const float* __restrict__ xs, const float* __restrict__ xr,
    const int* __restrict__ sidx, const int* __restrict__ ridx,
    const float* __restrict__ inv, int ninv,
    const float* __restrict__ W1, const float* __restrict__ b1,
    const float* __restrict__ W2, const float* __restrict__ b2,
    const float* __restrict__ W3, const float* __restrict__ b3,
    float* __restrict__ agg, int E, int act1, int act2) {
  __shared__ __align__(16) bf16_t sW1f[5 * 4 * 32 * 16];   // fragment-major, K padded to 160
  __shared__ __align__(16) bf16_t sW2f[2 * 4 * 32 * 16];
  __shared__ float sb1[64], sb2[64], sW3[64];
  __shared__ __align__(16) bf16_t sM[WAVES_PER_BLOCK][16 * 64];

  const int tid = threadIdx.x;
  swizzle_weights(sW1f, W1, 5, 128 + ninv, tid);
  swizzle_weights(sW2f, W2, 2, 64, tid);
  if (tid < 64) { sb1[tid] = b1[tid]; sb2[tid] = b2[tid]; sW3[tid] = W3[tid]; }
  __syncthreads();

  const float b3v = b3[0];
  const int wave = tid >> 5, lane = tid & 31;
  const int half = lane >> 4, lrow = lane & 15;
  bf16_t* mstage = sM[wave];
  const int ntiles = (E + 15) >> 4;
  const int tstride = (int)gridDim.x * WAVES_PER_BLOCK;

  for (int tile = (int)blockIdx.x * WAVES_PER_BLOCK + wave; tile < ntiles; tile += tstride) {
    const int e0 = tile * 16;
    const int myrow = e0 + lrow;
    const int crow = (myrow < E) ? myrow : (E - 1);
    const int si = sidx[crow];
    const int ri = ridx[crow];
    if (tile + tstride < ntiles) {  // speculative prefetch of next tile's indices
      __builtin_prefetch(sidx + (size_t)(tile + tstride) * 16 + lrow, 0, 1);
      __builtin_prefetch(ridx + (size_t)(tile + tstride) * 16 + lrow, 0, 1);
    }
    const float* prs = xs + (size_t)si * 64;
    const float* prr = xr + (size_t)ri * 64;
    const float* pri = inv + (size_t)crow * ninv;

    // ---- layer 1: h @ W1 -------------------------------------------------
    v8f acc[4];
#pragma unroll
    for (int t = 0; t < 4; ++t) acc[t] = vzero8();
#pragma unroll
    for (int kc = 0; kc < 5; ++kc) {
      v16bf a;
      if (kc < 2)       a = afrag_row64(prs, kc * 32, half);
      else if (kc < 4)  a = afrag_row64(prr, (kc - 2) * 32, half);
      else              a = afrag_inv(pri, ninv, half);
#pragma unroll
      for (int nt = 0; nt < 4; ++nt) {
        v16bf b = bfrag(sW1f, kc, nt, lane);
        acc[nt] = WMMA_BF16(a, b, acc[nt]);
      }
    }
    // bias + act1, stage to LDS for re-use as A of layer 2
#pragma unroll
    for (int nt = 0; nt < 4; ++nt) {
      const float bn = sb1[nt * 16 + lrow];
#pragma unroll
      for (int r = 0; r < 8; ++r) {
        float v = act_apply(acc[nt][r] + bn, act1);
        mstage[(half * 8 + r) * 64 + nt * 16 + lrow] = (bf16_t)v;
      }
    }
    asm volatile("s_wait_dscnt 0x0" ::: "memory");  // cross-lane LDS hand-off within wave

    // ---- layer 2: m @ W2 -------------------------------------------------
    v8f acc2[4];
#pragma unroll
    for (int t = 0; t < 4; ++t) acc2[t] = vzero8();
#pragma unroll
    for (int kc = 0; kc < 2; ++kc) {
      v16bf a = afrag_stage(mstage, kc, lane);
#pragma unroll
      for (int nt = 0; nt < 4; ++nt) {
        v16bf b = bfrag(sW2f, kc, nt, lane);
        acc2[nt] = WMMA_BF16(a, b, acc2[nt]);
      }
    }
#pragma unroll
    for (int nt = 0; nt < 4; ++nt) {
      const float bn = sb2[nt * 16 + lrow];
#pragma unroll
      for (int r = 0; r < 8; ++r) acc2[nt][r] = act_apply(acc2[nt][r] + bn, act2);
    }

    // ---- gate: per-row reduction m2 @ W3 --------------------------------
    float ps[8];
#pragma unroll
    for (int r = 0; r < 8; ++r) ps[r] = 0.0f;
#pragma unroll
    for (int nt = 0; nt < 4; ++nt) {
      const float w3n = sW3[nt * 16 + lrow];
#pragma unroll
      for (int r = 0; r < 8; ++r) ps[r] += acc2[nt][r] * w3n;
    }
#pragma unroll
    for (int m = 1; m < 16; m <<= 1) {
#pragma unroll
      for (int r = 0; r < 8; ++r) ps[r] += __shfl_xor(ps[r], m, 32);
    }

    // ---- scatter: agg[recv] += m2 * gate --------------------------------
#pragma unroll
    for (int r = 0; r < 8; ++r) {
      const int row = half * 8 + r;
      const int rr = __shfl(ri, row, 32);  // recv index of row, held by lane==row
      if (e0 + row < E) {
        const float g = sigm(ps[r] + b3v);
        float* dst = agg + (size_t)rr * 64 + lrow;
#pragma unroll
        for (int nt = 0; nt < 4; ++nt) atomicAdd(dst + nt * 16, acc2[nt][r] * g);
      }
    }
  }
}

// ---------------------------------------------------------------------------
// Node update: out = sigmoid(aggA) [+ sigmoid(aggB)];
//              y   = x + sigmoid(out@U1+c1)@U2 + c2
// ---------------------------------------------------------------------------
__device__ __forceinline__ v16bf afrag_out(const float* __restrict__ pa,
                                           const float* __restrict__ pb,
                                           int kc, int half) {
  const v4f* qa = (const v4f*)(pa + kc * 32 + half * 8);
  v4f a0 = qa[0], a1 = qa[1], a2 = qa[4], a3 = qa[5];
  float t[16];
#pragma unroll
  for (int j = 0; j < 4; ++j) {
    t[j]      = sigm(a0[j]);
    t[4 + j]  = sigm(a1[j]);
    t[8 + j]  = sigm(a2[j]);
    t[12 + j] = sigm(a3[j]);
  }
  if (pb) {
    const v4f* qb = (const v4f*)(pb + kc * 32 + half * 8);
    v4f b0 = qb[0], b1 = qb[1], b2 = qb[4], b3 = qb[5];
#pragma unroll
    for (int j = 0; j < 4; ++j) {
      t[j]      += sigm(b0[j]);
      t[4 + j]  += sigm(b1[j]);
      t[8 + j]  += sigm(b2[j]);
      t[12 + j] += sigm(b3[j]);
    }
  }
  v16bf a;
#pragma unroll
  for (int e = 0; e < 16; ++e) a[e] = (bf16_t)t[e];
  return a;
}

__global__ __launch_bounds__(TPB) void node_update_kernel(
    const float* __restrict__ x,
    const float* __restrict__ aggA, const float* __restrict__ aggB,
    const float* __restrict__ U1, const float* __restrict__ c1,
    const float* __restrict__ U2, const float* __restrict__ c2,
    float* __restrict__ y, int N) {
  __shared__ __align__(16) bf16_t sU1f[2 * 4 * 32 * 16];
  __shared__ __align__(16) bf16_t sU2f[2 * 4 * 32 * 16];
  __shared__ float sc1[64], sc2[64];
  __shared__ __align__(16) bf16_t sT[WAVES_PER_BLOCK][16 * 64];

  const int tid = threadIdx.x;
  swizzle_weights(sU1f, U1, 2, 64, tid);
  swizzle_weights(sU2f, U2, 2, 64, tid);
  if (tid < 64) { sc1[tid] = c1[tid]; sc2[tid] = c2[tid]; }
  __syncthreads();

  const int wave = tid >> 5, lane = tid & 31;
  const int half = lane >> 4, lrow = lane & 15;
  bf16_t* tstage = sT[wave];
  const int ntiles = (N + 15) >> 4;
  const int tstride = (int)gridDim.x * WAVES_PER_BLOCK;

  for (int tile = (int)blockIdx.x * WAVES_PER_BLOCK + wave; tile < ntiles; tile += tstride) {
    const int n0 = tile * 16;
    const int myrow = n0 + lrow;
    const int crow = (myrow < N) ? myrow : (N - 1);
    const float* pa = aggA + (size_t)crow * 64;
    const float* pb = aggB ? aggB + (size_t)crow * 64 : nullptr;

    v8f acc[4];
#pragma unroll
    for (int t = 0; t < 4; ++t) acc[t] = vzero8();
#pragma unroll
    for (int kc = 0; kc < 2; ++kc) {
      v16bf a = afrag_out(pa, pb, kc, half);
#pragma unroll
      for (int nt = 0; nt < 4; ++nt) {
        v16bf b = bfrag(sU1f, kc, nt, lane);
        acc[nt] = WMMA_BF16(a, b, acc[nt]);
      }
    }
#pragma unroll
    for (int nt = 0; nt < 4; ++nt) {
      const float cn = sc1[nt * 16 + lrow];
#pragma unroll
      for (int r = 0; r < 8; ++r)
        tstage[(half * 8 + r) * 64 + nt * 16 + lrow] = (bf16_t)sigm(acc[nt][r] + cn);
    }
    asm volatile("s_wait_dscnt 0x0" ::: "memory");

    v8f acc2[4];
#pragma unroll
    for (int t = 0; t < 4; ++t) acc2[t] = vzero8();
#pragma unroll
    for (int kc = 0; kc < 2; ++kc) {
      v16bf a = afrag_stage(tstage, kc, lane);
#pragma unroll
      for (int nt = 0; nt < 4; ++nt) {
        v16bf b = bfrag(sU2f, kc, nt, lane);
        acc2[nt] = WMMA_BF16(a, b, acc2[nt]);
      }
    }
#pragma unroll
    for (int r = 0; r < 8; ++r) {
      const int row = half * 8 + r;
      const int node = n0 + row;
      if (node < N) {
#pragma unroll
        for (int nt = 0; nt < 4; ++nt) {
          const int col = nt * 16 + lrow;
          y[(size_t)node * 64 + col] = x[(size_t)node * 64 + col] + acc2[nt][r] + sc2[col];
        }
      }
    }
  }
}

static inline int clamp_blocks(int b) { return b > MAX_BLOCKS ? MAX_BLOCKS : b; }

extern "C" void kernel_launch(void* const* d_in, const int* in_sizes, int n_in,
                              void* d_out, int out_size, void* d_ws, size_t ws_size,
                              hipStream_t stream) {
  const float* x0 = (const float*)d_in[0];
  const float* x1 = (const float*)d_in[1];
  const float* x2 = (const float*)d_in[2];
  const float* inv00 = (const float*)d_in[3];
  const float* inv11 = (const float*)d_in[4];
  const float* inv01 = (const float*)d_in[5];
  const float* inv12 = (const float*)d_in[6];

  const int N0 = in_sizes[0] / 64, N1 = in_sizes[1] / 64, N2 = in_sizes[2] / 64;
  const int EA0 = in_sizes[7], EA1 = in_sizes[9], EI1 = in_sizes[11], EI2 = in_sizes[13];
  const int ni00 = in_sizes[3] / EA0, ni11 = in_sizes[4] / EA1;
  const int ni01 = in_sizes[5] / EI1, ni12 = in_sizes[6] / EI2;

  auto p = [&](int i) { return (const float*)d_in[i]; };
  auto ip = [&](int i) { return (const int*)d_in[i]; };
  // params flat order: same0{W1,b1,W2,b2,W3,b3}=15..20, same1=21..26,
  //                    l2h1=27..32, l2h2=33..38, upd0{U1,c1,U2,c2}=39..42,
  //                    upd1=43..46, upd2=47..50

  float* ws = (float*)d_ws;
  float* agg0  = ws;
  float* agg1a = agg0 + (size_t)N0 * 64;
  float* agg1b = agg1a + (size_t)N1 * 64;
  float* agg2  = agg1b + (size_t)N1 * 64;
  const size_t aggBytes = ((size_t)N0 + 2 * (size_t)N1 + (size_t)N2) * 64 * sizeof(float);
  hipMemsetAsync(d_ws, 0, aggBytes, stream);

  auto blocksE = [](int E) {
    int nt = (E + 15) / 16;
    return clamp_blocks((nt + WAVES_PER_BLOCK - 1) / WAVES_PER_BLOCK);
  };

  // same-rank blocks (act1=sigmoid, act2=sigmoid)
  edge_msg_kernel<<<blocksE(EA0), TPB, 0, stream>>>(
      x0, x0, ip(7), ip(8), inv00, ni00,
      p(15), p(16), p(17), p(18), p(19), p(20), agg0, EA0, 1, 1);
  edge_msg_kernel<<<blocksE(EA1), TPB, 0, stream>>>(
      x1, x1, ip(9), ip(10), inv11, ni11,
      p(21), p(22), p(23), p(24), p(25), p(26), agg1a, EA1, 1, 1);
  // low-to-high blocks (act1=identity, act2=silu)
  edge_msg_kernel<<<blocksE(EI1), TPB, 0, stream>>>(
      x0, x1, ip(11), ip(12), inv01, ni01,
      p(27), p(28), p(29), p(30), p(31), p(32), agg1b, EI1, 0, 2);
  edge_msg_kernel<<<blocksE(EI2), TPB, 0, stream>>>(
      x1, x2, ip(13), ip(14), inv12, ni12,
      p(33), p(34), p(35), p(36), p(37), p(38), agg2, EI2, 0, 2);

  float* y0 = (float*)d_out;
  float* y1 = y0 + (size_t)N0 * 64;
  float* y2 = y1 + (size_t)N1 * 64;
  auto blocksN = [](int N) {
    int nt = (N + 15) / 16;
    return clamp_blocks((nt + WAVES_PER_BLOCK - 1) / WAVES_PER_BLOCK);
  };

  node_update_kernel<<<blocksN(N0), TPB, 0, stream>>>(
      x0, agg0, nullptr, p(39), p(40), p(41), p(42), y0, N0);
  node_update_kernel<<<blocksN(N1), TPB, 0, stream>>>(
      x1, agg1a, agg1b, p(43), p(44), p(45), p(46), y1, N1);
  node_update_kernel<<<blocksN(N2), TPB, 0, stream>>>(
      x2, agg2, nullptr, p(47), p(48), p(49), p(50), y2, N2);
}
